// VMambaUnet_4406636445793
// MI455X (gfx1250) — compile-verified, matrix operand order checked by reading, not run on hardware
//
#include <hip/hip_runtime.h>

typedef __attribute__((ext_vector_type(16))) _Float16 v16h;
typedef __attribute__((ext_vector_type(2)))  _Float16 v2h;
typedef __attribute__((ext_vector_type(8)))  float    v8f;

static inline int    cdiv(int a, int b)   { return (a + b - 1) / b; }
static inline size_t alignup(size_t b)    { return (b + 255) & ~(size_t)255; }

// ============================ device kernels ============================

// Block tile: 32x32 C, 4 waves (one 16x16 quadrant each), K staged 32 at a time.
// LDS: sA[m][k] f16 (row pad 8 halfs), sB[n][k] f16 (transposed at stage time).
// WMMA 16-bit A-fragment layout makes each lane's 16 halves two contiguous
// 16-byte runs: [8*hh .. 8*hh+7] and [16+8*hh .. 23+8*hh]  ->  ds_load_b128 x2.
// Interior tiles take a uniform branch-free staging path (float2 loads, packed
// b32 DS stores); only edge tiles run the per-element guarded path.
#define GT_MN 32
#define GT_K  32
#define TSTR  40   // 32 + 8 halfs pad: 80B row stride, conflict-free b128 reads

template<bool ACCUM>
__global__ __launch_bounds__(128)
void gemm_wmma_f16(const float* __restrict__ A, int lda,
                   const float* __restrict__ B, int ldb,
                   float* __restrict__ C, int ldc,
                   int M, int N, int K) {
  __shared__ _Float16 sA[GT_MN * TSTR];
  __shared__ _Float16 sB[GT_MN * TSTR];

  const int tid  = threadIdx.x;
  const int wave = tid >> 5;          // 0..3
  const int lane = tid & 31;
  const int hh   = lane >> 4;         // K-half selector / C row-half
  const int lan  = lane & 15;
  const int tileM0 = blockIdx.y * GT_MN;
  const int tileN0 = blockIdx.x * GT_MN;
  const int wm = (wave >> 1) * 16;    // quadrant offsets within block tile
  const int wn = (wave & 1) * 16;
  const int cm = tileM0 + wm;
  const int ccol = tileN0 + wn + lan;
  const bool interior = (tileM0 + GT_MN <= M) && (tileN0 + GT_MN <= N);

  v8f acc;
  if (ACCUM && interior) {
#pragma unroll
    for (int i = 0; i < 8; ++i)
      acc[i] = C[(size_t)(cm + i + 8 * hh) * ldc + ccol];
  } else {
#pragma unroll
    for (int i = 0; i < 8; ++i) {
      int mi = cm + i + 8 * hh;
      acc[i] = (ACCUM && mi < M && ccol < N) ? C[(size_t)mi * ldc + ccol] : 0.0f;
    }
  }

  for (int k0 = 0; k0 < K; k0 += GT_K) {
    if (interior && (k0 + GT_K <= K)) {
      // ---- fast path: unconditional, coalesced, vectorized ----
#pragma unroll
      for (int i = 0; i < 4; ++i) {
        int e = tid + i * 128;          // 0..511 float2 slots
        int r = e >> 4, c2 = e & 15;    // row, float2 column
        float2 v = *(const float2*)(A + (size_t)(tileM0 + r) * lda + k0 + 2 * c2);
        v2h h2; h2[0] = (_Float16)v.x; h2[1] = (_Float16)v.y;
        *(v2h*)(sA + r * TSTR + 2 * c2) = h2;
      }
#pragma unroll
      for (int i = 0; i < 4; ++i) {
        int e = tid + i * 128;
        int r = e >> 4, c2 = e & 15;    // r = k, c2 = n pair
        float2 v = *(const float2*)(B + (size_t)(k0 + r) * ldb + tileN0 + 2 * c2);
        sB[(2 * c2 + 0) * TSTR + r] = (_Float16)v.x;   // transposed store
        sB[(2 * c2 + 1) * TSTR + r] = (_Float16)v.y;
      }
    } else {
      // ---- edge path: per-element guarded, zero-filled ----
#pragma unroll
      for (int i = 0; i < 8; ++i) {
        int e = tid + i * 128;
        int r = e >> 5, c = e & 31;
        int gm = tileM0 + r, gk = k0 + c;
        float v = (gm < M && gk < K) ? A[(size_t)gm * lda + gk] : 0.0f;
        sA[r * TSTR + c] = (_Float16)v;
      }
#pragma unroll
      for (int i = 0; i < 8; ++i) {
        int e = tid + i * 128;
        int r = e >> 5, c = e & 31;
        int gk = k0 + r, gn = tileN0 + c;
        float v = (gk < K && gn < N) ? B[(size_t)gk * ldb + gn] : 0.0f;
        sB[c * TSTR + r] = (_Float16)v;
      }
    }
    __syncthreads();

    v16h a, b;
    const _Float16* pa = &sA[(wm + lan) * TSTR + 8 * hh];
    const _Float16* pb = &sB[(wn + lan) * TSTR + 8 * hh];
#pragma unroll
    for (int j = 0; j < 8; ++j) { a[j] = pa[j]; a[8 + j] = pa[16 + j]; }
#pragma unroll
    for (int j = 0; j < 8; ++j) { b[j] = pb[j]; b[8 + j] = pb[16 + j]; }

    acc = __builtin_amdgcn_wmma_f32_16x16x32_f16(false, a, false, b,
                                                 (short)0, acc, false, false);
    __syncthreads();
  }

  if (interior) {
#pragma unroll
    for (int i = 0; i < 8; ++i)
      C[(size_t)(cm + i + 8 * hh) * ldc + ccol] = acc[i];
  } else {
#pragma unroll
    for (int i = 0; i < 8; ++i) {
      int mi = cm + i + 8 * hh;
      if (mi < M && ccol < N) C[(size_t)mi * ldc + ccol] = acc[i];
    }
  }
}

// LayerNorm over last dim C; one 256-thread block per row; in-place safe.
__global__ void layernorm_k(const float* __restrict__ x, float* __restrict__ y,
                            const float* __restrict__ g, const float* __restrict__ b,
                            int C) {
  __shared__ float red[256];
  const int row = blockIdx.x;
  const float* xr = x + (size_t)row * C;
  float* yr = y + (size_t)row * C;
  float s = 0.f;
  for (int c = threadIdx.x; c < C; c += 256) s += xr[c];
  red[threadIdx.x] = s; __syncthreads();
  for (int st = 128; st > 0; st >>= 1) {
    if ((int)threadIdx.x < st) red[threadIdx.x] += red[threadIdx.x + st];
    __syncthreads();
  }
  const float mu = red[0] / C;
  __syncthreads();
  float v = 0.f;
  for (int c = threadIdx.x; c < C; c += 256) { float d = xr[c] - mu; v += d * d; }
  red[threadIdx.x] = v; __syncthreads();
  for (int st = 128; st > 0; st >>= 1) {
    if ((int)threadIdx.x < st) red[threadIdx.x] += red[threadIdx.x + st];
    __syncthreads();
  }
  const float rstd = rsqrtf(red[0] / C + 1e-5f);
  for (int c = threadIdx.x; c < C; c += 256)
    yr[c] = (xr[c] - mu) * rstd * g[c] + b[c];
}

// Patch embed: x NCHW (3,224,224), w (4,4,3,48) HWIO, stride 4 -> out [3136,48]
__global__ void patch_embed_k(const float* __restrict__ x, const float* __restrict__ w,
                              const float* __restrict__ b, float* __restrict__ out) {
  int idx = blockIdx.x * 256 + threadIdx.x;
  if (idx >= 56 * 56 * 48) return;
  int o = idx % 48, l = idx / 48;
  int wx = l % 56, hy = l / 56;
  float acc = b[o];
  for (int ky = 0; ky < 4; ++ky)
    for (int kx = 0; kx < 4; ++kx)
      for (int ci = 0; ci < 3; ++ci)
        acc += x[ci * 224 * 224 + (hy * 4 + ky) * 224 + (wx * 4 + kx)] *
               w[((ky * 4 + kx) * 3 + ci) * 48 + o];
  out[idx] = acc;
}

// Depthwise 3x3 SAME + bias + SiLU. Input strided (xc = first D cols of xz).
__global__ void dwconv_silu_k(const float* __restrict__ xin, int stride,
                              const float* __restrict__ w, const float* __restrict__ b,
                              float* __restrict__ out, int H, int W, int D) {
  int idx = blockIdx.x * 256 + threadIdx.x;
  if (idx >= H * W * D) return;
  int d = idx % D, l = idx / D;
  int wx = l % W, hy = l / W;
  float acc = b[d];
  for (int ky = -1; ky <= 1; ++ky)
    for (int kx = -1; kx <= 1; ++kx) {
      int yy = hy + ky, xx = wx + kx;
      if (yy >= 0 && yy < H && xx >= 0 && xx < W)
        acc += xin[(size_t)(yy * W + xx) * stride + d] *
               w[((ky + 1) * 3 + (kx + 1)) * D + d];
    }
  out[idx] = acc / (1.f + __expf(-acc));
}

// Build the 4 scan orders: xs[k][l][d]
__global__ void build_xs_k(const float* __restrict__ xc, float* __restrict__ xs,
                           int H, int W, int D) {
  int L = H * W;
  int idx = blockIdx.x * 256 + threadIdx.x;
  if (idx >= L * D) return;
  int d = idx % D, l = idx / D;
  int wx = l % W, hy = l / W;
  int lt = wx * H + hy;           // column-major position
  float v = xc[idx];
  xs[(size_t)(0 * L + l) * D + d]            = v;
  xs[(size_t)(1 * L + lt) * D + d]           = v;
  xs[(size_t)(2 * L + (L - 1 - l)) * D + d]  = v;
  xs[(size_t)(3 * L + (L - 1 - lt)) * D + d] = v;
}

// dt = softplus(dt + dt_bias[k][d]), in place; shape [4,L,D]
__global__ void softplus_bias_k(float* __restrict__ dt, const float* __restrict__ bias,
                                int L, int D) {
  int idx = blockIdx.x * 256 + threadIdx.x;
  if (idx >= 4 * L * D) return;
  int d = idx % D;
  int k = idx / (L * D);
  float x = dt[idx] + bias[k * D + d];
  dt[idx] = (x > 20.f) ? x : log1pf(__expf(x));
}

// Selective scan: one thread per (k,d), 16-element state in registers.
// proj rows: [dt(R) | B(16) | C(16)], stride P. ys[k][l][d] = y + xs*Ds.
__global__ void scan_k(const float* __restrict__ dt, const float* __restrict__ xs,
                       const float* __restrict__ proj, int P, int R,
                       const float* __restrict__ A_log, const float* __restrict__ Ds,
                       float* __restrict__ ys, int L, int D) {
  int t = blockIdx.x * 256 + threadIdx.x;
  if (t >= 4 * D) return;
  int k = t / D, d = t % D;
  float A[16], h[16];
#pragma unroll
  for (int n = 0; n < 16; ++n) {
    A[n] = -__expf(A_log[((size_t)k * D + d) * 16 + n]);
    h[n] = 0.f;
  }
  const float dsv = Ds[k * D + d];
  const float* dtp = dt + (size_t)k * L * D + d;
  const float* xp  = xs + (size_t)k * L * D + d;
  const float* pp  = proj + (size_t)k * L * P + R;
  float* yp = ys + (size_t)k * L * D + d;
  for (int l = 0; l < L; ++l) {
    float dtv = dtp[(size_t)l * D];
    float xv  = xp[(size_t)l * D];
    const float* pb = pp + (size_t)l * P;   // B at pb[0..15], C at pb[16..31]
    float dtx = dtv * xv, y = 0.f;
#pragma unroll
    for (int n = 0; n < 16; ++n) {
      h[n] = __expf(dtv * A[n]) * h[n] + dtx * pb[n];
      y += h[n] * pb[16 + n];
    }
    yp[(size_t)l * D] = y + xv * dsv;
  }
}

// Combine 4 directions: y[l] = ys0[l] + ys2[L-1-l] + ys1[lt] + ys3[L-1-lt]
__global__ void combine_k(const float* __restrict__ ys, float* __restrict__ y,
                          int H, int W, int D) {
  int L = H * W;
  int idx = blockIdx.x * 256 + threadIdx.x;
  if (idx >= L * D) return;
  int d = idx % D, l = idx / D;
  int wx = l % W, hy = l / W;
  int lt = wx * H + hy;
  y[idx] = ys[(size_t)(0 * L + l) * D + d] +
           ys[(size_t)(2 * L + (L - 1 - l)) * D + d] +
           ys[(size_t)(1 * L + lt) * D + d] +
           ys[(size_t)(3 * L + (L - 1 - lt)) * D + d];
}

// y *= silu(z); z strided in xz at column offset D (row stride = stride)
__global__ void silu_mul_k(float* __restrict__ y, const float* __restrict__ xz,
                           int stride, int D, int total) {
  int idx = blockIdx.x * 256 + threadIdx.x;
  if (idx >= total) return;
  int d = idx % D, l = idx / D;
  float z = xz[(size_t)l * stride + D + d];
  y[idx] *= z / (1.f + __expf(-z));
}

// Swin PatchMerging gather: [H,W,C] -> [H/2*W/2, 4C]
__global__ void merge_gather_k(const float* __restrict__ y, float* __restrict__ out,
                               int H, int W, int C) {
  int H2 = H / 2, W2 = W / 2, C4 = 4 * C;
  int total = H2 * W2 * C4;
  int idx = blockIdx.x * 256 + threadIdx.x;
  if (idx >= total) return;
  int c4 = idx % C4, l = idx / C4;
  int wx = l % W2, hy = l / W2;
  int q = c4 / C, c = c4 % C;
  int dy = (q == 1 || q == 3) ? 1 : 0;
  int dx = (q >= 2) ? 1 : 0;
  out[idx] = y[(size_t)((2 * hy + dy) * W + (2 * wx + dx)) * C + c];
}

// PatchExpand x2: t [H*W, 4c] -> out [(2H)*(2W), c]
__global__ void expand2_k(const float* __restrict__ t, float* __restrict__ out,
                          int H, int W, int c) {
  int total = 4 * H * W * c;
  int idx = blockIdx.x * 256 + threadIdx.x;
  if (idx >= total) return;
  int cc = idx % c, l = idx / c;
  int ox = l % (2 * W), oy = l / (2 * W);
  int h = oy >> 1, r = oy & 1, w = ox >> 1, s = ox & 1;
  out[idx] = t[(size_t)(h * W + w) * (4 * c) + (r * 2 + s) * c + cc];
}

// FinalPatchExpand x4: t [H*W, 16c] -> out [(4H)*(4W), c]
__global__ void expand4_k(const float* __restrict__ t, float* __restrict__ out,
                          int H, int W, int c) {
  int total = 16 * H * W * c;
  int idx = blockIdx.x * 256 + threadIdx.x;
  if (idx >= total) return;
  int cc = idx % c, l = idx / c;
  int ox = l % (4 * W), oy = l / (4 * W);
  int h = oy >> 2, r = oy & 3, w = ox >> 2, s = ox & 3;
  out[idx] = t[(size_t)(h * W + w) * (16 * c) + (r * 4 + s) * c + cc];
}

// out[l, 0:C] = a[l], out[l, C:2C] = b[l]
__global__ void concat_k(const float* __restrict__ a, const float* __restrict__ bsrc,
                         float* __restrict__ out, int L, int C) {
  int total = L * 2 * C;
  int idx = blockIdx.x * 256 + threadIdx.x;
  if (idx >= total) return;
  int c2 = idx % (2 * C), l = idx / (2 * C);
  out[idx] = (c2 < C) ? a[(size_t)l * C + c2] : bsrc[(size_t)l * C + (c2 - C)];
}

// 1x1 conv head with bhwc,ck->bkhw transpose: out[k*L + l]
__global__ void head_k(const float* __restrict__ y, const float* __restrict__ w,
                       float* __restrict__ out, int L) {
  int idx = blockIdx.x * 256 + threadIdx.x;
  if (idx >= L * 9) return;
  int kc = idx % 9, l = idx / 9;
  float acc = 0.f;
  for (int c = 0; c < 96; ++c) acc += y[(size_t)l * 96 + c] * w[c * 9 + kc];
  out[(size_t)kc * L + l] = acc;
}

// ============================ host side ============================

struct Blk {
  const float *ng, *nb;
  const float *in_proj, *conv_w, *conv_b, *x_proj, *dt_proj, *dt_bias,
              *A_log, *Ds, *ln_g, *ln_b, *out_proj;
};

static void gemm(const float* A, int lda, const float* B, int ldb,
                 float* C, int ldc, int M, int N, int K, bool accum,
                 hipStream_t st) {
  dim3 grid(cdiv(N, GT_MN), cdiv(M, GT_MN));
  if (accum) gemm_wmma_f16<true ><<<grid, 128, 0, st>>>(A, lda, B, ldb, C, ldc, M, N, K);
  else       gemm_wmma_f16<false><<<grid, 128, 0, st>>>(A, lda, B, ldb, C, ldc, M, N, K);
}

// One VSS block: y (L x C, in/out) += SS2D(LN(y))
static void run_block(const Blk& p, float* y, int H, int W, int C,
                      char* ws, size_t off, hipStream_t st) {
  auto sal = [&](size_t nf) -> float* {
    float* r = (float*)(ws + off); off += alignup(nf * 4); return r;
  };
  const int L = H * W, D = 2 * C;
  const int R = (C / 16) ? (C / 16) : 1;
  const int P = R + 32;   // R + 2*N_STATE

  float* xln = sal((size_t)L * C);
  layernorm_k<<<L, 256, 0, st>>>(y, xln, p.ng, p.nb, C);

  float* xz = sal((size_t)L * 2 * D);
  gemm(xln, C, p.in_proj, 2 * D, xz, 2 * D, L, 2 * D, C, false, st);

  float* xc = sal((size_t)L * D);
  dwconv_silu_k<<<cdiv(L * D, 256), 256, 0, st>>>(xz, 2 * D, p.conv_w, p.conv_b, xc, H, W, D);

  float* xs = sal((size_t)4 * L * D);
  build_xs_k<<<cdiv(L * D, 256), 256, 0, st>>>(xc, xs, H, W, D);

  float* proj = sal((size_t)4 * L * P);
  for (int k = 0; k < 4; ++k)
    gemm(xs + (size_t)k * L * D, D, p.x_proj + (size_t)k * D * P, P,
         proj + (size_t)k * L * P, P, L, P, D, false, st);

  float* dt = sal((size_t)4 * L * D);
  for (int k = 0; k < 4; ++k)
    gemm(proj + (size_t)k * L * P, P, p.dt_proj + (size_t)k * R * D, D,
         dt + (size_t)k * L * D, D, L, D, R, false, st);
  softplus_bias_k<<<cdiv(4 * L * D, 256), 256, 0, st>>>(dt, p.dt_bias, L, D);

  float* ys = sal((size_t)4 * L * D);
  scan_k<<<cdiv(4 * D, 256), 256, 0, st>>>(dt, xs, proj, P, R, p.A_log, p.Ds, ys, L, D);

  float* yc = sal((size_t)L * D);
  combine_k<<<cdiv(L * D, 256), 256, 0, st>>>(ys, yc, H, W, D);
  layernorm_k<<<L, 256, 0, st>>>(yc, yc, p.ln_g, p.ln_b, D);
  silu_mul_k<<<cdiv(L * D, 256), 256, 0, st>>>(yc, xz, 2 * D, D, L * D);

  // residual: y += yc @ out_proj
  gemm(yc, D, p.out_proj, C, y, C, L, C, D, true, st);
}

extern "C" void kernel_launch(void* const* d_in, const int* in_sizes, int n_in,
                              void* d_out, int out_size, void* d_ws, size_t ws_size,
                              hipStream_t stream) {
  (void)in_sizes; (void)n_in; (void)out_size; (void)ws_size;
  int pi = 0;
  auto nf = [&]() -> const float* { return (const float*)d_in[pi++]; };
  auto loadBlk = [&]() -> Blk {
    Blk b;
    b.ng = nf(); b.nb = nf();
    b.in_proj = nf(); b.conv_w = nf(); b.conv_b = nf(); b.x_proj = nf();
    b.dt_proj = nf(); b.dt_bias = nf(); b.A_log = nf(); b.Ds = nf();
    b.ln_g = nf(); b.ln_b = nf(); b.out_proj = nf();
    return b;
  };

  // ---- parameters, in setup_inputs() insertion order ----
  const float* ximg = nf();
  const float* pe_w = nf(); const float* pe_cb = nf();
  const float* pe_g = nf(); const float* pe_b  = nf();
  const float* embed_proj = nf();

  Blk dblk[3][2]; const float *mg[3], *mb[3], *mw[3];
  for (int s = 0; s < 3; ++s) {
    dblk[s][0] = loadBlk(); dblk[s][1] = loadBlk();
    mg[s] = nf(); mb[s] = nf(); mw[s] = nf();
  }
  Blk midb0 = loadBlk();
  Blk midb1 = loadBlk();
  const float *ew[3], *eg[3], *eb[3], *fw[3];
  Blk ublk[3][2];
  for (int s = 0; s < 3; ++s) {
    ew[s] = nf(); eg[s] = nf(); eb[s] = nf(); fw[s] = nf();
    ublk[s][0] = loadBlk(); ublk[s][1] = loadBlk();
  }
  const float* fin_w = nf(); const float* fin_g = nf(); const float* fin_b = nf();
  const float* head_w = nf();

  // ---- workspace layout ----
  char* ws = (char*)d_ws;
  size_t off = 0;
  auto pal = [&](size_t nfl) -> float* {
    float* r = (float*)(ws + off); off += alignup(nfl * 4); return r;
  };
  float* bufA = pal((size_t)50176 * 96);   // persistent ping
  float* bufB = pal((size_t)50176 * 96);   // persistent pong
  float* skips[3];
  skips[0] = pal((size_t)3136 * 96);
  skips[1] = pal((size_t)784 * 192);
  skips[2] = pal((size_t)196 * 384);
  const size_t scratch0 = off;             // per-step scratch, reused

  // ---- patch embed: 224x224x3 -> 56x56x48 -> LN -> @embed_proj -> 56x56x96 ----
  {
    size_t o = scratch0;
    float* t0 = (float*)(ws + o); o += alignup((size_t)3136 * 48 * 4);
    patch_embed_k<<<cdiv(3136 * 48, 256), 256, 0, stream>>>(ximg, pe_w, pe_cb, t0);
    layernorm_k<<<3136, 256, 0, stream>>>(t0, t0, pe_g, pe_b, 48);
    gemm(t0, 48, embed_proj, 96, bufA, 96, 3136, 96, 48, false, stream);
  }

  float* cur = bufA; float* oth = bufB;
  int H = 56, W = 56, C = 96;

  // ---- encoder ----
  for (int s = 0; s < 3; ++s) {
    run_block(dblk[s][0], cur, H, W, C, ws, scratch0, stream);
    run_block(dblk[s][1], cur, H, W, C, ws, scratch0, stream);
    hipMemcpyAsync(skips[s], cur, (size_t)H * W * C * sizeof(float),
                   hipMemcpyDeviceToDevice, stream);
    int H2 = H / 2, W2 = W / 2, L2 = H2 * W2;
    size_t o = scratch0;
    float* cat = (float*)(ws + o); o += alignup((size_t)L2 * 4 * C * 4);
    merge_gather_k<<<cdiv(L2 * 4 * C, 256), 256, 0, stream>>>(cur, cat, H, W, C);
    layernorm_k<<<L2, 256, 0, stream>>>(cat, cat, mg[s], mb[s], 4 * C);
    gemm(cat, 4 * C, mw[s], 2 * C, oth, 2 * C, L2, 2 * C, 4 * C, false, stream);
    float* t = cur; cur = oth; oth = t;
    H = H2; W = W2; C *= 2;
  }

  // ---- bottleneck ----
  run_block(midb0, cur, H, W, C, ws, scratch0, stream);
  run_block(midb1, cur, H, W, C, ws, scratch0, stream);

  // ---- decoder ----
  for (int s = 0; s < 3; ++s) {
    int L = H * W, Cout = C / 2;
    size_t o = scratch0;
    float* t = (float*)(ws + o); o += alignup((size_t)L * 2 * C * 4);
    gemm(cur, C, ew[s], 2 * C, t, 2 * C, L, 2 * C, C, false, stream);
    expand2_k<<<cdiv(4 * L * Cout, 256), 256, 0, stream>>>(t, oth, H, W, Cout);
    layernorm_k<<<4 * L, 256, 0, stream>>>(oth, oth, eg[s], eb[s], Cout);
    float* cat = (float*)(ws + o); o += alignup((size_t)4 * L * 2 * Cout * 4);
    concat_k<<<cdiv(4 * L * 2 * Cout, 256), 256, 0, stream>>>(oth, skips[2 - s], cat, 4 * L, Cout);
    gemm(cat, 2 * Cout, fw[s], Cout, cur, Cout, 4 * L, Cout, 2 * Cout, false, stream);
    H *= 2; W *= 2; C = Cout;
    run_block(ublk[s][0], cur, H, W, C, ws, scratch0, stream);
    run_block(ublk[s][1], cur, H, W, C, ws, scratch0, stream);
  }

  // ---- final x4 expand + LN + 1x1 head ----
  {
    size_t o = scratch0;
    float* t = (float*)(ws + o); o += alignup((size_t)3136 * 1536 * 4);
    gemm(cur, 96, fin_w, 1536, t, 1536, 3136, 1536, 96, false, stream);
    expand4_k<<<cdiv(50176 * 96, 256), 256, 0, stream>>>(t, oth, 56, 56, 96);
    layernorm_k<<<50176, 256, 0, stream>>>(oth, oth, fin_g, fin_b, 96);
    head_k<<<cdiv(50176 * 9, 256), 256, 0, stream>>>(oth, head_w, (float*)d_out, 50176);
  }
}
